// NetVLADLayer_37280316129457
// MI455X (gfx1250) — matrix operand order repaired and checked by reading, not compiled, not run on hardware
//
#include <hip/hip_runtime.h>

#define B_      32
#define HW_     1600
#define D_      512
#define K_      64
#define TILE_P  64
#define NTILES  (HW_ / TILE_P)            /* 25 */
#define VN      ((size_t)B_ * K_ * D_)    /* 1,048,576 floats */
#define XPAD    16                        /* pad x-tile rows: stride 528 floats */
#define XSTR    (D_ + XPAD)

typedef __attribute__((ext_vector_type(16))) __bf16 v16bf;
typedef __attribute__((ext_vector_type(8)))  __bf16 v8bf;
typedef __attribute__((ext_vector_type(8)))  float  v8f;
typedef __attribute__((ext_vector_type(2)))  float  v2f;

// ------------------------------------------------------------------ setup:
// zero v-accumulator + a_sum + gn, and build Wt = W^T in bf16 (K x D, d-contig)
__global__ void nv_setup(const float* __restrict__ W, float* __restrict__ zf,
                         __bf16* __restrict__ Wt, size_t nz) {
  size_t i = (size_t)blockIdx.x * blockDim.x + threadIdx.x;
  if (i < nz) zf[i] = 0.0f;
  if (i < (size_t)K_ * D_) {
    int k = (int)(i / D_), d = (int)(i % D_);
    Wt[i] = (__bf16)W[(size_t)d * K_ + k];
  }
}

// ------------------------------------------------------------------ fused:
// per (image, 64-position tile):
//   async-stage x tile -> LDS (ASYNCcnt, overlapped with matmul1)
//   s^T = Wt*x^T (bf16 WMMA) -> softmax in LDS -> a_sum
//   v += a^T * x (f32 WMMA 16x16x4, B-frags from LDS) with global f32 atomics.
__global__ __launch_bounds__(256) void nv_main(
    const float* __restrict__ x, const __bf16* __restrict__ Wt,
    const float* __restrict__ bias, float* __restrict__ vacc,
    float* __restrict__ a_sum) {
  __shared__ float xs[TILE_P][XSTR];       // 64 x 528 f32 staged x tile (135 KB)
  __shared__ float sA[TILE_P][K_ + 1];     // 64 x 65 f32: logits, then softmax'd a

  const int tid  = threadIdx.x;
  const int lane = tid & 31;
  const int wave = tid >> 5;
  const int b    = blockIdx.x / NTILES;
  const int tile = blockIdx.x % NTILES;
  const float* xb = x + ((size_t)b * HW_ + (size_t)tile * TILE_P) * D_;

  // ============ async stage: x tile (64x512 f32) -> xs, tracked by ASYNCcnt ==
  {
    unsigned lds0 = (unsigned)(size_t)(&xs[0][0]);   // flat -> LDS byte offset
#pragma unroll
    for (int i = 0; i < 32; ++i) {
      int e    = i * 256 + tid;          // 16-byte chunk id, 0..8191
      int row  = e >> 7;                 // 128 chunks per 512-float row
      int c4   = (e & 127) << 2;         // float column
      unsigned     lds = lds0 + (unsigned)(row * XSTR + c4) * 4u;
      const float* g   = xb + (size_t)row * D_ + c4;
      asm volatile("global_load_async_to_lds_b128 %0, %1, off"
                   :: "v"(lds), "v"(g) : "memory");
    }
  }

  // ============ matmul1: s^T(64x64) = Wt(64x512,bf16) . x^T (global x) ======
  {
    const int pt  = (wave & 3) * 16;       // position tile base
    const int c0  = (wave >> 2) * 32;      // this wave owns 2 cluster tiles
    const int l15 = lane & 15;
    const int hi  = lane >> 4;             // 0 | 1 (lane half)

    v8f acc0 = {}; v8f acc1 = {};
    const float* xrow = xb + (size_t)(pt + l15) * D_;   // B-frag row (pos)

    for (int d0 = 0; d0 < D_; d0 += 32) {
      // B fragment: x[pos][d0 + 16*hi .. +15]  fp32 -> bf16 (16 contiguous)
      const float4* xp = (const float4*)(xrow + d0 + 16 * hi);
      float4 q0 = xp[0], q1 = xp[1], q2 = xp[2], q3 = xp[3];
      v16bf bf;
      bf[0]=(__bf16)q0.x;  bf[1]=(__bf16)q0.y;  bf[2]=(__bf16)q0.z;  bf[3]=(__bf16)q0.w;
      bf[4]=(__bf16)q1.x;  bf[5]=(__bf16)q1.y;  bf[6]=(__bf16)q1.z;  bf[7]=(__bf16)q1.w;
      bf[8]=(__bf16)q2.x;  bf[9]=(__bf16)q2.y;  bf[10]=(__bf16)q2.z; bf[11]=(__bf16)q2.w;
      bf[12]=(__bf16)q3.x; bf[13]=(__bf16)q3.y; bf[14]=(__bf16)q3.z; bf[15]=(__bf16)q3.w;

      const int dA = d0 + 16 * hi;         // A k order (same as B): K {0..15}/{16..31} by lane half
      {
        const v8bf* wp = (const v8bf*)(Wt + (size_t)(c0 + l15) * D_ + dA);
        v8bf lo = wp[0], hh = wp[1];       // +8 bf16 elements
        v16bf af = __builtin_shufflevector(lo, hh, 0,1,2,3,4,5,6,7,8,9,10,11,12,13,14,15);
        acc0 = __builtin_amdgcn_wmma_f32_16x16x32_bf16(false, af, false, bf,
                                                       (short)0, acc0, false, false);
      }
      {
        const v8bf* wp = (const v8bf*)(Wt + (size_t)(c0 + 16 + l15) * D_ + dA);
        v8bf lo = wp[0], hh = wp[1];
        v16bf af = __builtin_shufflevector(lo, hh, 0,1,2,3,4,5,6,7,8,9,10,11,12,13,14,15);
        acc1 = __builtin_amdgcn_wmma_f32_16x16x32_bf16(false, af, false, bf,
                                                       (short)0, acc1, false, false);
      }
    }
    // C layout: lanes0-15 N=pos=lane, VGPR r -> M = r (+8 for lanes 16-31)
#pragma unroll
    for (int r = 0; r < 8; ++r) {
      int c = c0 + r + 8 * hi;
      sA[pt + l15][c]      = acc0[r] + bias[c];
      sA[pt + l15][c + 16] = acc1[r] + bias[c + 16];
    }
  }
  __syncthreads();

  // ============ softmax over K per position (in LDS) ============
  if (tid < TILE_P) {
    float m = -1e30f;
#pragma unroll 4
    for (int c = 0; c < K_; ++c) m = fmaxf(m, sA[tid][c]);
    float s = 0.0f;
#pragma unroll 4
    for (int c = 0; c < K_; ++c) { float e = __expf(sA[tid][c] - m); sA[tid][c] = e; s += e; }
    float inv = 1.0f / s;
#pragma unroll 4
    for (int c = 0; c < K_; ++c) sA[tid][c] *= inv;
  }
  // staging must be complete (and visible to all waves) past this barrier
  asm volatile("s_wait_asynccnt 0x0" ::: "memory");
  __syncthreads();

  // a_sum[b][k] += sum_p a[p][k]
  if (tid < K_) {
    float s = 0.0f;
#pragma unroll 4
    for (int p = 0; p < TILE_P; ++p) s += sA[p][tid];
    atomicAdd(a_sum + b * K_ + tid, s);
  }

  // ============ matmul2: v(64x512) += a^T(64x64) . x(64x512), f32 WMMA ====
  {
    const int mc    = (wave & 3) * 16;     // cluster tile
    const int dbase = (wave >> 2) * 256;   // channel half
    const int l15   = lane & 15;
    const int hi2   = (lane >> 4) * 2;     // K offset for upper lanes

    v8f vt[16];
#pragma unroll
    for (int i = 0; i < 16; ++i) vt[i] = (v8f){};

    for (int ps = 0; ps < TILE_P; ps += 4) {
      // A frag (16x4 f32): lane m holds a^T[mc+m][ps + j + hi2], j=0,1
      v2f af;
      af[0] = sA[ps + 0 + hi2][mc + l15];
      af[1] = sA[ps + 1 + hi2][mc + l15];
      const float* xr0 = &xs[ps + 0 + hi2][dbase + l15];
      const float* xr1 = xr0 + XSTR;
#pragma unroll
      for (int nt = 0; nt < 16; ++nt) {
        // B frag (4x16 f32): lane n holds x[ps + j + hi2][dbase + nt*16 + n]
        v2f bf;
        bf[0] = xr0[nt * 16];
        bf[1] = xr1[nt * 16];
        vt[nt] = __builtin_amdgcn_wmma_f32_16x16x4_f32(false, af, false, bf,
                                                       (short)0, vt[nt], false, false);
      }
    }
    // accumulate into global v[b][c][d]
    float* vb = vacc + (size_t)b * K_ * D_;
#pragma unroll
    for (int nt = 0; nt < 16; ++nt) {
      int d = dbase + nt * 16 + l15;
#pragma unroll
      for (int r = 0; r < 8; ++r) {
        int c = mc + r + 8 * (lane >> 4);
        atomicAdd(vb + (size_t)c * D_ + d, vt[nt][r]);
      }
    }
  }
}

// ------------------------------------------------------------------ finalize 1:
// per (b,k): w = v + a_sum*C^T, intra-normalize over D, write out, collect gn.
__global__ __launch_bounds__(128) void nv_fin1(
    const float* __restrict__ vacc, const float* __restrict__ a_sum,
    const float* __restrict__ C, float* __restrict__ out, float* __restrict__ gn) {
  __shared__ float red[4];
  const int blk = blockIdx.x;
  const int b = blk >> 6, k = blk & 63;
  const int tid = threadIdx.x;
  const float* vr = vacc + ((size_t)b * K_ + k) * D_;
  const float as = a_sum[b * K_ + k];

  float w[4]; float ss = 0.0f;
#pragma unroll
  for (int i = 0; i < 4; ++i) {
    int d = tid + i * 128;
    w[i] = vr[d] + as * C[(size_t)d * K_ + k];
    ss += w[i] * w[i];
  }
  for (int m = 16; m; m >>= 1) ss += __shfl_xor(ss, m, 32);
  if ((tid & 31) == 0) red[tid >> 5] = ss;
  __syncthreads();
  float tot = red[0] + red[1] + red[2] + red[3];
  float inv = 1.0f / fmaxf(sqrtf(tot), 1e-12f);
  float* orow = out + ((size_t)b * K_ + k) * D_;
#pragma unroll
  for (int i = 0; i < 4; ++i) orow[tid + i * 128] = w[i] * inv;
  if (tid == 0) atomicAdd(gn + b, tot * inv * inv);   // = ||normalized row||^2
}

// ------------------------------------------------------------------ finalize 2:
__global__ void nv_fin2(float* __restrict__ out, const float* __restrict__ gn) {
  const int b = blockIdx.x;
  const float s = 1.0f / fmaxf(sqrtf(gn[b]), 1e-12f);
  float* orow = out + (size_t)b * (K_ * D_);
  for (int i = threadIdx.x; i < K_ * D_; i += blockDim.x) orow[i] *= s;
}

// ------------------------------------------------------------------ launch
extern "C" void kernel_launch(void* const* d_in, const int* in_sizes, int n_in,
                              void* d_out, int out_size, void* d_ws, size_t ws_size,
                              hipStream_t stream) {
  const float* x    = (const float*)d_in[0];
  const float* W    = (const float*)d_in[1];
  const float* bias = (const float*)d_in[2];
  const float* C    = (const float*)d_in[3];
  float* out = (float*)d_out;

  float*  wsf   = (float*)d_ws;
  float*  vacc  = wsf;                       // VN floats
  float*  a_sum = wsf + VN;                  // B*K floats
  float*  gn    = a_sum + (size_t)B_ * K_;   // B floats
  __bf16* Wt    = (__bf16*)(gn + B_);        // K*D bf16 (16B-aligned offset)

  const size_t nz = VN + (size_t)B_ * K_ + B_;

  nv_setup<<<(unsigned)((nz + 255) / 256), 256, 0, stream>>>(W, wsf, Wt, nz);
  nv_main<<<B_ * NTILES, 256, 0, stream>>>(x, Wt, bias, vacc, a_sum);
  nv_fin1<<<B_ * K_, 128, 0, stream>>>(vacc, a_sum, C, out, gn);
  nv_fin2<<<B_, 256, 0, stream>>>(out, gn);
}
